// GroupSelfAttention_53730040873216
// MI455X (gfx1250) — matrix operand order repaired and checked by reading, not compiled, not run on hardware
//
#include <hip/hip_runtime.h>

// ---------------------------------------------------------------------------
// GroupSelfAttention for MI455X (gfx1250, wave32, WMMA f32_16x16x32_f16)
//
// Pipeline (all intermediates carved from d_ws):
//   1. convert weights / x to f16 (K-contiguous layouts)
//   2. embed MLP  -> E[v,ij][kl(pad208)][64]  (row_e|col_e concat, f16)
//   3. QKV GEMM (WMMA)   -> qT/kT [bh][784][96], vF [bh][96][800], qpos[ij][64][64]
//   4. content GEMM (WMMA): per (b,h) 784x784x96 f32
//   5. pos GEMM (WMMA): per (v,ij) 196x64x64 -> pos_s f32
//   6. g_s scalar kernel
//   7. softmax over (m,kl)=784, summed over g -> Pbar f16 [bhv][208][800]
//   8. vo GEMM (WMMA): per (b,h,v) 96x196x800 -> voT f16 [b,v,ij][768]
//   9. out GEMM (WMMA): 96x1568x768 + bias -> d_out f32
//
// All GEMMs use 32x32 register blocking per wave (2x2 WMMA tiles, 4 accums):
// 2 A + 2 B fragment loads feed 4 v_wmma per k-step. Store epilogues fully
// unrolled (constant accumulator indexing, no v_movrels).
// ---------------------------------------------------------------------------

typedef _Float16 f16;
typedef __attribute__((ext_vector_type(16))) _Float16 v16h;
typedef __attribute__((ext_vector_type(8)))  float    v8f;

// ---- wave-level 32x32 macro-tile GEMM: A (MxK row-major), Bt (NxK row-major)
// Fragment layouts per CDNA5 ISA 7.12.2 (wave32):
//   A 16-bit 16x32: lane<16 holds M=lane, K = {k0..k0+7, k0+16..k0+23}
//                   lane>=16 holds M=lane-16, K = {k0+8..k0+15, k0+24..k0+31}
//   B 16-bit 32x16: lane holds N=lane&15, K same as A (k0+8*(lane>=16) +{0..7,16..23})
//   C f32 16x16: VGPR r -> M = r + 8*(lane>=16), N = lane&15
__device__ __forceinline__ void wmma32x32(
    const f16* __restrict__ A, int lda, int mA, int mB,
    const f16* __restrict__ Bt, int ldb, int nA, int nB,
    int K, int lane, v8f& c00, v8f& c01, v8f& c10, v8f& c11) {
  const int ln = lane & 15;
  const int hi = (lane >> 4) & 1;
  const f16* pa0 = A + (size_t)(mA + ln) * lda + hi * 8;
  const f16* pa1 = A + (size_t)(mB + ln) * lda + hi * 8;
  const f16* pb0 = Bt + (size_t)(nA + ln) * ldb + hi * 8;
  const f16* pb1 = Bt + (size_t)(nB + ln) * ldb + hi * 8;
#pragma unroll
  for (int k0 = 0; k0 < K; k0 += 32) {
    __builtin_prefetch(pa0 + k0 + 64, 0, 1);
    __builtin_prefetch(pb0 + k0 + 64, 0, 1);
    union { v16h v; uint4 u[2]; } a0, a1, bb0, bb1;
    a0.u[0] = *(const uint4*)(pa0 + k0);
    a0.u[1] = *(const uint4*)(pa0 + k0 + 16);
    a1.u[0] = *(const uint4*)(pa1 + k0);
    a1.u[1] = *(const uint4*)(pa1 + k0 + 16);
    bb0.u[0] = *(const uint4*)(pb0 + k0);
    bb0.u[1] = *(const uint4*)(pb0 + k0 + 16);
    bb1.u[0] = *(const uint4*)(pb1 + k0);
    bb1.u[1] = *(const uint4*)(pb1 + k0 + 16);
    v16h b0 = bb0.v;
    v16h b1 = bb1.v;
    c00 = __builtin_amdgcn_wmma_f32_16x16x32_f16(false, a0.v, false, b0, (short)0, c00, false, false);
    c01 = __builtin_amdgcn_wmma_f32_16x16x32_f16(false, a0.v, false, b1, (short)0, c01, false, false);
    c10 = __builtin_amdgcn_wmma_f32_16x16x32_f16(false, a1.v, false, b0, (short)0, c10, false, false);
    c11 = __builtin_amdgcn_wmma_f32_16x16x32_f16(false, a1.v, false, b1, (short)0, c11, false, false);
  }
}

// ---------------- conversion kernels ----------------
__global__ __launch_bounds__(256) void k_conv_w(
    const float* __restrict__ Wq, const float* __restrict__ Wk,
    const float* __restrict__ Wv, const float* __restrict__ Wo,
    f16* __restrict__ Wf, f16* __restrict__ WoF) {
  int idx = blockIdx.x * 256 + threadIdx.x;           // 294912 threads
  if (idx < 73728)        Wf[idx] = (f16)Wq[idx];
  else if (idx < 147456)  Wf[idx] = (f16)Wk[idx - 73728];
  else if (idx < 221184)  Wf[idx] = (f16)Wv[idx - 147456];
  else                    WoF[idx - 221184] = (f16)Wo[idx - 221184];
}

__global__ __launch_bounds__(256) void k_conv_x(const float* __restrict__ x,
                                                f16* __restrict__ xT) {
  int idx = blockIdx.x * 256 + threadIdx.x;           // 150528 threads
  int c = idx % 96, p = idx / 96;
  int b = p / 784, rem = p % 784;                     // rem = g*196+ij
  xT[idx] = (f16)x[((size_t)b * 96 + c) * 784 + rem];
}

// ---------------- embed MLP ----------------
__device__ __forceinline__ void embed_one(float rel,
    const float* w1, const float* b1, const float* gam, const float* bet,
    const float* w2, const float* b2, f16* dst) {
  float h[16];
  float mu = 0.f;
#pragma unroll
  for (int t = 0; t < 16; ++t) { h[t] = rel * w1[t] + b1[t]; mu += h[t]; }
  mu *= (1.f / 16.f);
  float var = 0.f;
#pragma unroll
  for (int t = 0; t < 16; ++t) { float d = h[t] - mu; var += d * d; }
  var *= (1.f / 16.f);
  float inv = rsqrtf(var + 1e-5f);
#pragma unroll
  for (int t = 0; t < 16; ++t) {
    float hn = (h[t] - mu) * inv * gam[t] + bet[t];
    h[t] = hn / (1.f + __expf(-hn));                  // silu
  }
  for (int c = 0; c < 32; ++c) {
    float o = b2[c];
#pragma unroll
    for (int t = 0; t < 16; ++t) o += h[t] * w2[t * 32 + c];
    dst[c] = (f16)o;
  }
}

__global__ __launch_bounds__(256) void k_embed(
    const float* __restrict__ row_rel, const float* __restrict__ col_rel,
    const float* rw1, const float* rb1, const float* rgam, const float* rbet,
    const float* rw2, const float* rb2,
    const float* cw1, const float* cb1, const float* cgam, const float* cbet,
    const float* cw2, const float* cb2, f16* __restrict__ E) {
  int idx = blockIdx.x * 256 + threadIdx.x;           // 784*208 = 163072
  int kl = idx % 208, vij = idx / 208;                // vij = v*196+ij
  f16* dst = E + ((size_t)vij * 208 + kl) * 64;
  if (kl >= 196) { for (int c = 0; c < 64; ++c) dst[c] = (f16)0.f; return; }
  int src = vij * 196 + kl;
  embed_one(row_rel[src], rw1, rb1, rgam, rbet, rw2, rb2, dst);
  embed_one(col_rel[src], cw1, cb1, cgam, cbet, cw2, cb2, dst + 32);
}

// ---------------- store helpers (fully unrolled epilogues) ----------------
__device__ __forceinline__ void store_q(const v8f& acc, int m0, int n0, int lane,
    const float* bias, f16* __restrict__ qT, f16* __restrict__ qpos) {
  int hi = (lane >> 4) & 1, ln = lane & 15;
  int n = n0 + ln;
  int b = n / 784, rem2 = n % 784;                    // rem2 = g*196+ij
  int g = rem2 / 196, ij = rem2 % 196;
#pragma unroll
  for (int r = 0; r < 8; ++r) {
    int d = m0 + r + 8 * hi;                          // 0..767
    float val = acc[r] + bias[d];
    int c = d >> 3, h = d & 7;                        // reshape (MID,HEADS)
    int bh = b * 8 + h;
    qT[((size_t)(bh * 784 + rem2)) * 96 + c] = (f16)val;
    if (c < 64) qpos[((size_t)(ij * 64 + bh * 4 + g)) * 64 + c] = (f16)val;
  }
}

__device__ __forceinline__ void store_k(const v8f& acc, int m0, int n0, int lane,
    const float* bias, f16* __restrict__ kT) {
  int hi = (lane >> 4) & 1, ln = lane & 15;
  int n = n0 + ln;
  int b = n / 784, rem2 = n % 784;
#pragma unroll
  for (int r = 0; r < 8; ++r) {
    int d = m0 + r + 8 * hi;
    float val = acc[r] + bias[d];
    int c = d >> 3, h = d & 7;
    kT[((size_t)((b * 8 + h) * 784 + rem2)) * 96 + c] = (f16)val;
  }
}

__device__ __forceinline__ void store_v(const v8f& acc, int m0, int n0, int lane,
    const float* bias, f16* __restrict__ vF) {
  int hi = (lane >> 4) & 1, ln = lane & 15;
  int n = n0 + ln;
  int b = n / 784, rem2 = n % 784;
#pragma unroll
  for (int r = 0; r < 8; ++r) {
    int d = m0 + r + 8 * hi;
    float val = acc[r] + bias[d];
    int c = d >> 3, h = d & 7;
    vF[((size_t)((b * 8 + h) * 96 + c)) * 800 + rem2] = (f16)val;
  }
}

__device__ __forceinline__ void store_tile_f32(const v8f& acc, float* __restrict__ out,
                                               int ld, int m0, int n0, int lane) {
  int hi = (lane >> 4) & 1, ln = lane & 15;
  int N = n0 + ln;
#pragma unroll
  for (int r = 0; r < 8; ++r)
    out[(size_t)(m0 + r + 8 * hi) * ld + N] = acc[r];
}

__device__ __forceinline__ void store_pos(const v8f& acc, float* __restrict__ pos_s,
                                          int v, int ij, int m0, int n0, int lane) {
  int hi = (lane >> 4) & 1, ln = lane & 15;
  int n = n0 + ln;                                    // n = b*32 + h*4 + g
  int b = n >> 5, h = (n >> 2) & 7, g = n & 3;
  int bh = b * 8 + h;
  float* dst = pos_s + ((size_t)((bh * 4 + v) * 4 + g) * 196 + ij) * 196;
#pragma unroll
  for (int r = 0; r < 8; ++r) {
    int kl = m0 + r + 8 * hi;
    if (kl < 196) dst[kl] = acc[r];
  }
}

__device__ __forceinline__ void store_vo(const v8f& acc, f16* __restrict__ voT,
                                         int b, int h, int v, int m0, int n0, int lane) {
  int hi = (lane >> 4) & 1, ln = lane & 15;
  int ij = n0 + ln;
  if (ij >= 196) return;
  f16* dst = voT + ((size_t)(b * 4 + v) * 196 + ij) * 768;
#pragma unroll
  for (int r = 0; r < 8; ++r) {
    int c = m0 + r + 8 * hi;
    dst[c * 8 + h] = (f16)acc[r];                     // channel = c*HEADS+h
  }
}

__device__ __forceinline__ void store_out(const v8f& acc, float* __restrict__ out,
                                          const float* bo, int m0, int n0, int lane) {
  int hi = (lane >> 4) & 1, ln = lane & 15;
  int n = n0 + ln;
  int b = n / 784, rem = n % 784;
  int v = rem / 196, ij = rem % 196;
#pragma unroll
  for (int r = 0; r < 8; ++r) {
    int d = m0 + r + 8 * hi;
    out[((size_t)(b * 96 + d) * 4 + v) * 196 + ij] = acc[r] + bo[d];
  }
}

// ---------------- QKV projection GEMM: M=768, N=1568, K=96 ----------------
__global__ __launch_bounds__(128) void k_gemm_qkv(
    const f16* __restrict__ Wf, const f16* __restrict__ xT,
    const float* __restrict__ bq, const float* __restrict__ bk,
    const float* __restrict__ bv,
    f16* __restrict__ qT, f16* __restrict__ kT,
    f16* __restrict__ vF, f16* __restrict__ qpos) {
  int w = blockIdx.x * 4 + (threadIdx.x >> 5);        // 3*24*49 = 3528 waves
  int lane = threadIdx.x & 31;
  int proj = w / (24 * 49);
  int rem = w % (24 * 49);
  int mb = rem / 49, nb = rem % 49;
  int mA = mb * 32, mB = mA + 16, nA = nb * 32, nB = nA + 16;
  const f16* A = Wf + (size_t)proj * 73728;
  v8f c00 = {}, c01 = {}, c10 = {}, c11 = {};
  wmma32x32(A, 96, mA, mB, xT, 96, nA, nB, 96, lane, c00, c01, c10, c11);
  if (proj == 0) {
    store_q(c00, mA, nA, lane, bq, qT, qpos);
    store_q(c01, mA, nB, lane, bq, qT, qpos);
    store_q(c10, mB, nA, lane, bq, qT, qpos);
    store_q(c11, mB, nB, lane, bq, qT, qpos);
  } else if (proj == 1) {
    store_k(c00, mA, nA, lane, bk, kT);
    store_k(c01, mA, nB, lane, bk, kT);
    store_k(c10, mB, nA, lane, bk, kT);
    store_k(c11, mB, nB, lane, bk, kT);
  } else {
    store_v(c00, mA, nA, lane, bv, vF);
    store_v(c01, mA, nB, lane, bv, vF);
    store_v(c10, mB, nA, lane, bv, vF);
    store_v(c11, mB, nB, lane, bv, vF);
  }
}

// ---------------- content = Q^T K: per (b,h) 784x784x96 ----------------
__global__ __launch_bounds__(128) void k_gemm_content(
    const f16* __restrict__ qT, const f16* __restrict__ kT,
    float* __restrict__ content) {
  int w = blockIdx.x * 4 + (threadIdx.x >> 5);        // 16*25*25 = 10000 waves
  int lane = threadIdx.x & 31;
  int bh = w / 625, rem = w % 625;
  int mb = rem / 25, nb = rem % 25;
  int mA = mb * 32, mB = min(mA + 16, 768);           // 784 rows: clamp last pair
  int nA = nb * 32, nB = min(nA + 16, 768);
  const f16* A = qT + (size_t)bh * 784 * 96;
  const f16* Bt = kT + (size_t)bh * 784 * 96;
  v8f c00 = {}, c01 = {}, c10 = {}, c11 = {};
  wmma32x32(A, 96, mA, mB, Bt, 96, nA, nB, 96, lane, c00, c01, c10, c11);
  float* out = content + (size_t)bh * 784 * 784;
  store_tile_f32(c00, out, 784, mA, nA, lane);
  store_tile_f32(c01, out, 784, mA, nB, lane);
  store_tile_f32(c10, out, 784, mB, nA, lane);
  store_tile_f32(c11, out, 784, mB, nB, lane);
}

// ---------------- pos scores: per (v,ij), 208x64x64 ----------------
__global__ __launch_bounds__(128) void k_gemm_pos(
    const f16* __restrict__ E, const f16* __restrict__ qpos,
    float* __restrict__ pos_s) {
  int w = blockIdx.x * 4 + (threadIdx.x >> 5);        // 784*14 = 10976 waves
  int lane = threadIdx.x & 31;
  int vij = w / 14, t = w % 14;
  int mb = t / 2, nb = t & 1;
  int v = vij / 196, ij = vij % 196;
  int mA = mb * 32, mB = min(mA + 16, 192);           // 13 tiles: clamp last pair
  int nA = nb * 32, nB = nA + 16;
  const f16* A = E + (size_t)vij * 208 * 64;
  const f16* Bt = qpos + (size_t)ij * 64 * 64;
  v8f c00 = {}, c01 = {}, c10 = {}, c11 = {};
  wmma32x32(A, 64, mA, mB, Bt, 64, nA, nB, 64, lane, c00, c01, c10, c11);
  store_pos(c00, pos_s, v, ij, mA, nA, lane);
  store_pos(c01, pos_s, v, ij, mA, nB, lane);
  store_pos(c10, pos_s, v, ij, mB, nA, lane);
  store_pos(c11, pos_s, v, ij, mB, nB, lane);
}

// ---------------- g_s (tiny) ----------------
__global__ __launch_bounds__(256) void k_gs(
    const f16* __restrict__ qT, const float* __restrict__ gemb,
    const int* __restrict__ g_idx, float* __restrict__ g_s) {
  int idx = blockIdx.x * 256 + threadIdx.x;           // 200704 threads
  int m = idx & 3, t = idx >> 2;
  int ij = t % 196; t /= 196;
  int g = t & 3; t >>= 2;
  int v = t & 3; int bh = t >> 2;
  int gi = g_idx[(v * 4 + g) * 4 + m];
  const f16* q = qT + ((size_t)(bh * 784 + g * 196 + ij)) * 96 + 64;
  const float* e = gemb + gi * 32;
  float s = 0.f;
#pragma unroll
  for (int c = 0; c < 32; ++c) s += (float)q[c] * e[c];
  g_s[idx] = s;
}

// ---------------- softmax over (m,kl)=784, summed over g ----------------
__global__ __launch_bounds__(256) void k_softmax(
    const float* __restrict__ content, const float* __restrict__ pos_s,
    const float* __restrict__ g_s, f16* __restrict__ Pbar) {
  __shared__ float red[256];
  int tid = threadIdx.x;
  int bid = blockIdx.x;                               // (bh*4+v)*196 + ij
  int ij = bid % 196, t = bid / 196;
  int v = t & 3, bh = t >> 2;
  const float inv_scale = 0.10206207261596575f;       // 1/sqrt(96)
  float acc4[4] = {0.f, 0.f, 0.f, 0.f};
  for (int g = 0; g < 4; ++g) {
    const float* crow = content + ((size_t)bh * 784 + g * 196 + ij) * 784;
    const float* prow = pos_s + ((size_t)((bh * 4 + v) * 4 + g) * 196 + ij) * 196;
    const float* grow = g_s + ((size_t)((bh * 4 + v) * 4 + g) * 196 + ij) * 4;
    float sc[4];
    float lmax = -3.0e38f;
#pragma unroll
    for (int j = 0; j < 4; ++j) {
      int idx = tid + 256 * j;
      if (idx < 784) {
        int m = idx / 196, kl = idx % 196;
        float s = (prow[kl] + grow[m] + crow[idx]) * inv_scale;
        sc[j] = s;
        lmax = fmaxf(lmax, s);
      }
    }
    red[tid] = lmax; __syncthreads();
    for (int s2 = 128; s2 > 0; s2 >>= 1) {
      if (tid < s2) red[tid] = fmaxf(red[tid], red[tid + s2]);
      __syncthreads();
    }
    float mx = red[0]; __syncthreads();
    float lsum = 0.f;
#pragma unroll
    for (int j = 0; j < 4; ++j) {
      int idx = tid + 256 * j;
      if (idx < 784) { float e = __expf(sc[j] - mx); sc[j] = e; lsum += e; }
    }
    red[tid] = lsum; __syncthreads();
    for (int s2 = 128; s2 > 0; s2 >>= 1) {
      if (tid < s2) red[tid] += red[tid + s2];
      __syncthreads();
    }
    float invs = 1.f / red[0]; __syncthreads();
#pragma unroll
    for (int j = 0; j < 4; ++j) {
      int idx = tid + 256 * j;
      if (idx < 784) acc4[j] += sc[j] * invs;
    }
  }
  f16* pb = Pbar + ((size_t)(bh * 4 + v) * 208 + ij) * 800;
#pragma unroll
  for (int j = 0; j < 4; ++j) {
    int idx = tid + 256 * j;
    if (idx < 800) pb[idx] = (f16)((idx < 784) ? acc4[j] : 0.f);
  }
}

// ---------------- vo = V x Pbar^T per (b,h,v): 96x208x800 ----------------
__global__ __launch_bounds__(128) void k_gemm_vo(
    const f16* __restrict__ vF, const f16* __restrict__ Pbar,
    f16* __restrict__ voT) {
  int w = blockIdx.x * 4 + (threadIdx.x >> 5);        // 64*21 = 1344 waves
  int lane = threadIdx.x & 31;
  int bhv = w / 21, t = w % 21;
  int mb = t / 7, nb = t % 7;
  int bh = bhv >> 2, v = bhv & 3;
  int b = bh >> 3, h = bh & 7;
  int mA = mb * 32, mB = mA + 16;                     // 96 = 3*32 exact
  int nA = nb * 32, nB = min(nA + 16, 192);           // 13 tiles: clamp last pair
  const f16* A = vF + (size_t)bh * 96 * 800;
  const f16* Bt = Pbar + (size_t)bhv * 208 * 800;
  v8f c00 = {}, c01 = {}, c10 = {}, c11 = {};
  wmma32x32(A, 800, mA, mB, Bt, 800, nA, nB, 800, lane, c00, c01, c10, c11);
  store_vo(c00, voT, b, h, v, mA, nA, lane);
  store_vo(c01, voT, b, h, v, mA, nB, lane);
  store_vo(c10, voT, b, h, v, mB, nA, lane);
  store_vo(c11, voT, b, h, v, mB, nB, lane);
}

// ---------------- output projection: 96x1568x768 ----------------
__global__ __launch_bounds__(96) void k_gemm_out(
    const f16* __restrict__ WoF, const f16* __restrict__ voT,
    const float* __restrict__ bo, float* __restrict__ out) {
  int mb = threadIdx.x >> 5;                          // 3 waves/block
  int nb = blockIdx.x;                                // 49 blocks
  int lane = threadIdx.x & 31;
  int mA = mb * 32, mB = mA + 16;                     // 96 = 3*32 exact
  int nA = nb * 32, nB = nA + 16;                     // 1568 = 49*32 exact
  v8f c00 = {}, c01 = {}, c10 = {}, c11 = {};
  wmma32x32(WoF, 768, mA, mB, voT, 768, nA, nB, 768, lane, c00, c01, c10, c11);
  store_out(c00, out, bo, mA, nA, lane);
  store_out(c01, out, bo, mA, nB, lane);
  store_out(c10, out, bo, mB, nA, lane);
  store_out(c11, out, bo, mB, nB, lane);
}

// ---------------------------------------------------------------------------
extern "C" void kernel_launch(void* const* d_in, const int* in_sizes, int n_in,
                              void* d_out, int out_size, void* d_ws, size_t ws_size,
                              hipStream_t stream) {
  (void)in_sizes; (void)n_in; (void)out_size; (void)ws_size;
  const float* x    = (const float*)d_in[0];
  const float* Wq   = (const float*)d_in[1];
  const float* bq   = (const float*)d_in[2];
  const float* Wk   = (const float*)d_in[3];
  const float* bk   = (const float*)d_in[4];
  const float* Wv   = (const float*)d_in[5];
  const float* bv   = (const float*)d_in[6];
  const float* Wo   = (const float*)d_in[7];
  const float* bo   = (const float*)d_in[8];
  const float* rw1  = (const float*)d_in[9];
  const float* rb1  = (const float*)d_in[10];
  const float* rgam = (const float*)d_in[11];
  const float* rbet = (const float*)d_in[12];
  const float* rw2  = (const float*)d_in[13];
  const float* rb2  = (const float*)d_in[14];
  const float* cw1  = (const float*)d_in[15];
  const float* cb1  = (const float*)d_in[16];
  const float* cgam = (const float*)d_in[17];
  const float* cbet = (const float*)d_in[18];
  const float* cw2  = (const float*)d_in[19];
  const float* cb2  = (const float*)d_in[20];
  const float* gemb = (const float*)d_in[21];
  const float* row_rel = (const float*)d_in[22];
  const float* col_rel = (const float*)d_in[23];
  const int*   g_idx   = (const int*)d_in[24];
  float* out = (float*)d_out;

  char* ws = (char*)d_ws;
  size_t off = 0;
  auto carve = [&](size_t bytes) -> void* {
    void* p = ws + off;
    off = (off + bytes + 255) & ~(size_t)255;
    return p;
  };

  f16*   Wf      = (f16*)carve((size_t)3 * 768 * 96 * sizeof(f16));
  f16*   WoF     = (f16*)carve((size_t)96 * 768 * sizeof(f16));
  f16*   xT      = (f16*)carve((size_t)1568 * 96 * sizeof(f16));
  f16*   E       = (f16*)carve((size_t)784 * 208 * 64 * sizeof(f16));
  f16*   qpos    = (f16*)carve((size_t)196 * 64 * 64 * sizeof(f16));
  f16*   qT      = (f16*)carve((size_t)16 * 784 * 96 * sizeof(f16));
  f16*   kT      = (f16*)carve((size_t)16 * 784 * 96 * sizeof(f16));
  f16*   vF      = (f16*)carve((size_t)16 * 96 * 800 * sizeof(f16));
  float* content = (float*)carve((size_t)16 * 784 * 784 * sizeof(float));
  float* pos_s   = (float*)carve((size_t)16 * 16 * 196 * 196 * sizeof(float));
  float* g_s     = (float*)carve((size_t)16 * 16 * 196 * 4 * sizeof(float));
  f16*   Pbar    = (f16*)carve((size_t)64 * 208 * 800 * sizeof(f16));
  f16*   voT     = (f16*)carve((size_t)8 * 196 * 768 * sizeof(f16));

  // zero vF so the K-pad (784..799) region is zero for the vo GEMM
  hipMemsetAsync(vF, 0, (size_t)16 * 96 * 800 * sizeof(f16), stream);

  k_conv_w<<<1152, 256, 0, stream>>>(Wq, Wk, Wv, Wo, Wf, WoF);
  k_conv_x<<<588, 256, 0, stream>>>(x, xT);
  k_embed<<<637, 256, 0, stream>>>(row_rel, col_rel,
                                   rw1, rb1, rgam, rbet, rw2, rb2,
                                   cw1, cb1, cgam, cbet, cw2, cb2, E);
  k_gemm_qkv<<<882, 128, 0, stream>>>(Wf, xT, bq, bk, bv, qT, kT, vF, qpos);
  k_gemm_content<<<2500, 128, 0, stream>>>(qT, kT, content);
  k_gemm_pos<<<2744, 128, 0, stream>>>(E, qpos, pos_s);
  k_gs<<<784, 256, 0, stream>>>(qT, gemb, g_idx, g_s);
  k_softmax<<<12544, 256, 0, stream>>>(content, pos_s, g_s, Pbar);
  k_gemm_vo<<<336, 128, 0, stream>>>(vF, Pbar, voT);
  k_gemm_out<<<49, 96, 0, stream>>>(WoF, voT, bo, out);
}
